// GraphModel_5188320494487
// MI455X (gfx1250) — matrix operand order repaired and checked by reading, not compiled, not run on hardware
//
#include <hip/hip_runtime.h>

// ---------------------------------------------------------------------------
// GraphNet block x5 on gfx1250: edge/node/global MLPs via v_wmma_f32_16x16x32_f16
// kt-outer / nt-inner GEMMs: each LDS A fragment is loaded once and fed to all
// N-tile accumulators back-to-back.
// ---------------------------------------------------------------------------

#define N_NODES 50000
#define E_TOT   800000
#define BG      16
#define NPG     3125
#define EPG     50000
#define FD      32
#define HD      64

typedef _Float16 v16h __attribute__((ext_vector_type(16)));
typedef _Float16 v8h  __attribute__((ext_vector_type(8)));
typedef float    v8f  __attribute__((ext_vector_type(8)));

// Build the 16-bit A-matrix fragment (16x32) for one k-tile from an LDS row.
// Lane layout (ISA 7.12.2): lanes 0-15: K = {0..7, 16..23}; lanes 16-31: K = {8..15, 24..31}.
__device__ __forceinline__ v16h lds_a_frag(const _Float16* rowp, int base) {
  v8h c0 = *(const v8h*)(rowp + base);
  v8h c1 = *(const v8h*)(rowp + base + 16);
  v16h a;
#pragma unroll
  for (int i = 0; i < 8; ++i) { a[i] = c0[i]; a[i + 8] = c1[i]; }
  return a;
}

__device__ __forceinline__ void cvt4(_Float16* d, float4 v) {
  d[0] = (_Float16)v.x; d[1] = (_Float16)v.y;
  d[2] = (_Float16)v.z; d[3] = (_Float16)v.w;
}
__device__ __forceinline__ void cvt4s(_Float16* d, float4 v, float s) {
  d[0] = (_Float16)(v.x * s); d[1] = (_Float16)(v.y * s);
  d[2] = (_Float16)(v.z * s); d[3] = (_Float16)(v.w * s);
}

// Layer 1: [16x128(f16)] @ [128x64] + bias, ReLU -> f16 tile in Hbase (row stride 72).
// kt-outer: one A frag feeds 4 consecutive WMMAs.
__device__ __forceinline__ void gemm1_relu(const _Float16* Arow, _Float16* Hbase,
                                           const _Float16* Bp, const float* bias,
                                           int lane) {
  const int m16 = lane & 15, hi = lane >> 4;
  const int kb = hi * 8, mb = hi * 8;
  v8f acc[4];
#pragma unroll
  for (int nt = 0; nt < 4; ++nt) {
    const float bv = bias[nt * 16 + m16];
#pragma unroll
    for (int r = 0; r < 8; ++r) acc[nt][r] = bv;
  }
#pragma unroll
  for (int kt = 0; kt < 4; ++kt) {
    v16h a = lds_a_frag(Arow, kt * 32 + kb);
#pragma unroll
    for (int nt = 0; nt < 4; ++nt) {
      v16h b = *(const v16h*)(Bp + ((size_t)(kt * 4 + nt) * 32 + lane) * 16);
      acc[nt] = __builtin_amdgcn_wmma_f32_16x16x32_f16(false, a, false, b,
                                                       (short)0, acc[nt], false, false);
    }
  }
#pragma unroll
  for (int nt = 0; nt < 4; ++nt)
#pragma unroll
    for (int r = 0; r < 8; ++r) {
      float v = acc[nt][r] > 0.f ? acc[nt][r] : 0.f;
      Hbase[(size_t)(mb + r) * 72 + nt * 16 + m16] = (_Float16)v;
    }
}

// Layer 2: both 16x16 output tiles of [16x64(f16)] @ [64x32] + bias.
__device__ __forceinline__ void gemm2_both(const _Float16* Hrow, const _Float16* Bp,
                                           const float* bias, int lane,
                                           v8f* acc0, v8f* acc1) {
  const int m16 = lane & 15, kb = (lane >> 4) * 8;
  v8f a0, a1;
  {
    const float b0 = bias[m16], b1 = bias[16 + m16];
#pragma unroll
    for (int r = 0; r < 8; ++r) { a0[r] = b0; a1[r] = b1; }
  }
#pragma unroll
  for (int kt = 0; kt < 2; ++kt) {
    v16h a = lds_a_frag(Hrow, kt * 32 + kb);
    v16h w0 = *(const v16h*)(Bp + ((size_t)(kt * 2 + 0) * 32 + lane) * 16);
    a0 = __builtin_amdgcn_wmma_f32_16x16x32_f16(false, a, false, w0,
                                                (short)0, a0, false, false);
    v16h w1 = *(const v16h*)(Bp + ((size_t)(kt * 2 + 1) * 32 + lane) * 16);
    a1 = __builtin_amdgcn_wmma_f32_16x16x32_f16(false, a, false, w1,
                                                (short)0, a1, false, false);
  }
  *acc0 = a0; *acc1 = a1;
}

// ---------------------------------------------------------------------------
// Weight packing: convert f32 [Kact x Nout] row-major into per-fragment f16
// layout so each lane's v16h B fragment is one contiguous 32B load.
// B layout (ISA 7.12.2): lanes 0-15 hold K=kt*32+{0..7,16..23}, lanes 16-31 K=kt*32+{8..15,24..31}.
__global__ void pack_b_kernel(const float* __restrict__ W, _Float16* __restrict__ out,
                              int Kact, int Nout, int NT) {
  const int idx = blockIdx.x * blockDim.x + threadIdx.x;
  const int i = idx & 15;
  const int lane = (idx >> 4) & 31;
  const int frag = idx >> 9;
  const int nt = frag % NT;
  const int kt = frag / NT;
  const int n = nt * 16 + (lane & 15);
  const int k = kt * 32 + (i & 8) * 2 + ((lane >> 4) & 1) * 8 + (i & 7);
  const float v = (k < Kact) ? W[(size_t)k * Nout + n] : 0.f;
  out[idx] = (_Float16)v;
}

__global__ void zero_kernel(float* __restrict__ p, int n) {
  int i = blockIdx.x * blockDim.x + threadIdx.x;
  if (i < n) p[i] = 0.f;
}

__global__ void degree_kernel(const int* __restrict__ col, float* __restrict__ deg) {
  int e = blockIdx.x * blockDim.x + threadIdx.x;
  if (e < E_TOT) atomicAdd(&deg[col[e]], 1.0f);
}

// ---------------------------------------------------------------------------
// Edge model: e_in = [x[row], x[col], e, y[b]] (K=128) -> H=64 -> F=32
// 128 edges / block, 16 edges / wave. Updates edge_attr in place and
// scatter-adds into per-node and per-graph sums.
__global__ __launch_bounds__(256) void edge_kernel(
    const float* __restrict__ x, float* __restrict__ eattr,
    const float* __restrict__ y,
    const int* __restrict__ rowI, const int* __restrict__ colI,
    const _Float16* __restrict__ We1p, const float* __restrict__ be1,
    const _Float16* __restrict__ We2p, const float* __restrict__ be2,
    float* __restrict__ e_sum_node, float* __restrict__ e_sum_graph) {
  __shared__ _Float16 As[8][16][136];
  __shared__ _Float16 Hs[8][16][72];
  const int t = threadIdx.x;
  const int w = t >> 5, lane = t & 31;

  // ---- gather (2 threads per edge, same wave as the consumer) ----
  {
    const int le = t >> 1;
    const int part = t & 1;
    const size_t e = (size_t)blockIdx.x * 128u + (unsigned)le;
    const int wg = le >> 4, m = le & 15;
    _Float16* dst = &As[wg][m][0];
    if (part == 0) {
      const int r = rowI[e];
      const int c = colI[e];
      const float4* xs = (const float4*)(x + (size_t)r * FD);
      const float4* xd = (const float4*)(x + (size_t)c * FD);
#pragma unroll
      for (int i = 0; i < 8; ++i) cvt4(dst + i * 4, xs[i]);
#pragma unroll
      for (int i = 0; i < 8; ++i) cvt4(dst + 32 + i * 4, xd[i]);
    } else {
      const int b = (int)(e / EPG);  // edges are within-graph, EPG per graph
      const float4* es = (const float4*)(eattr + e * FD);
      const float4* ys = (const float4*)(y + (size_t)b * FD);
#pragma unroll
      for (int i = 0; i < 8; ++i) cvt4(dst + 64 + i * 4, es[i]);
#pragma unroll
      for (int i = 0; i < 8; ++i) cvt4(dst + 96 + i * 4, ys[i]);
    }
  }
  // wave-local LDS producer/consumer: DS ops are in-order per wave, no barrier

  const int m16 = lane & 15, hi = lane >> 4, mb = hi * 8;
  gemm1_relu(&As[w][m16][0], &Hs[w][0][0], We1p, be1, lane);

  const size_t e0 = (size_t)blockIdx.x * 128u + (unsigned)(w * 16);
  const int bg = (int)(e0 / EPG);  // uniform: 16 | EPG
  int cids[8];
#pragma unroll
  for (int r = 0; r < 8; ++r) cids[r] = colI[e0 + mb + r];

  v8f acc[2];
  gemm2_both(&Hs[w][m16][0], We2p, be2, lane, &acc[0], &acc[1]);
#pragma unroll
  for (int nt = 0; nt < 2; ++nt) {
    const int n = nt * 16 + m16;
    float gsum = 0.f;
#pragma unroll
    for (int r = 0; r < 8; ++r) {
      const float v = acc[nt][r];
      eattr[(e0 + mb + r) * FD + n] = v;
      atomicAdd(&e_sum_node[(size_t)cids[r] * FD + n], v);
      gsum += v;
    }
    gsum += __shfl_xor(gsum, 16);
    if (lane < 16) atomicAdd(&e_sum_graph[bg * FD + n], gsum);
  }
}

// ---------------------------------------------------------------------------
// Node model: n_in = [x, e_agg, y[b], 0pad] (K=96->128) -> H=64 -> F=32
__global__ __launch_bounds__(256) void node_kernel(
    float* __restrict__ x, const float* __restrict__ y,
    const float* __restrict__ e_sum_node, const float* __restrict__ deg,
    const _Float16* __restrict__ Wn1p, const float* __restrict__ bn1,
    const _Float16* __restrict__ Wn2p, const float* __restrict__ bn2,
    float* __restrict__ x_sum_graph) {
  __shared__ _Float16 As[8][16][136];
  __shared__ _Float16 Hs[8][16][72];
  const int t = threadIdx.x;
  const int w = t >> 5, lane = t & 31;

  {
    const int le = t >> 1;
    const int part = t & 1;
    const int n = blockIdx.x * 128 + le;
    if (n < N_NODES) {
      const int wg = le >> 4, m = le & 15;
      _Float16* dst = &As[wg][m][0];
      if (part == 0) {
        const float4* xs = (const float4*)(x + (size_t)n * FD);
#pragma unroll
        for (int i = 0; i < 8; ++i) cvt4(dst + i * 4, xs[i]);
        const float invd = 1.0f / fmaxf(deg[n], 1.0f);
        const float4* es = (const float4*)(e_sum_node + (size_t)n * FD);
#pragma unroll
        for (int i = 0; i < 8; ++i) cvt4s(dst + 32 + i * 4, es[i], invd);
      } else {
        const int b = n / NPG;
        const float4* ys = (const float4*)(y + (size_t)b * FD);
#pragma unroll
        for (int i = 0; i < 8; ++i) cvt4(dst + 64 + i * 4, ys[i]);
#pragma unroll
        for (int j = 0; j < 32; ++j) dst[96 + j] = (_Float16)0.f;
      }
    }
  }

  const int n0 = blockIdx.x * 128 + w * 16;
  if (n0 >= N_NODES) return;  // 16 | N_NODES: active waves fully valid

  const int m16 = lane & 15, hi = lane >> 4, mb = hi * 8;
  gemm1_relu(&As[w][m16][0], &Hs[w][0][0], Wn1p, bn1, lane);

  const int b0 = n0 / NPG, b15 = (n0 + 15) / NPG;
  v8f acc[2];
  gemm2_both(&Hs[w][m16][0], Wn2p, bn2, lane, &acc[0], &acc[1]);
#pragma unroll
  for (int nt = 0; nt < 2; ++nt) {
    const int n = nt * 16 + m16;
#pragma unroll
    for (int r = 0; r < 8; ++r) x[(size_t)(n0 + mb + r) * FD + n] = acc[nt][r];
    if (b0 == b15) {
      float gsum = 0.f;
#pragma unroll
      for (int r = 0; r < 8; ++r) gsum += acc[nt][r];
      gsum += __shfl_xor(gsum, 16);
      if (lane < 16) atomicAdd(&x_sum_graph[b0 * FD + n], gsum);
    } else {  // tile straddles a graph boundary (rare: 15 of 3125 tiles)
#pragma unroll
      for (int r = 0; r < 8; ++r)
        atomicAdd(&x_sum_graph[((n0 + mb + r) / NPG) * FD + n], acc[nt][r]);
    }
  }
}

// ---------------------------------------------------------------------------
// Global model: g_in = [x_agg, e_agg, u, 0pad] (16 rows) -> H=64 -> F=32
__global__ __launch_bounds__(32) void global_kernel(
    float* __restrict__ y,
    const float* __restrict__ x_sum_graph, const float* __restrict__ e_sum_graph,
    const _Float16* __restrict__ Wg1p, const float* __restrict__ bg1,
    const _Float16* __restrict__ Wg2p, const float* __restrict__ bg2) {
  __shared__ _Float16 As[16][136];
  __shared__ _Float16 Hs[16][72];
  const int lane = threadIdx.x;
  {
    const int g = lane >> 1, part = lane & 1;
    _Float16* dst = &As[g][0];
    if (part == 0) {
      const float4* xs = (const float4*)(x_sum_graph + (size_t)g * FD);
      const float4* es = (const float4*)(e_sum_graph + (size_t)g * FD);
#pragma unroll
      for (int i = 0; i < 8; ++i) cvt4s(dst + i * 4, xs[i], 1.0f / (float)NPG);
#pragma unroll
      for (int i = 0; i < 8; ++i) cvt4s(dst + 32 + i * 4, es[i], 1.0f / (float)EPG);
    } else {
      const float4* ys = (const float4*)(y + (size_t)g * FD);
#pragma unroll
      for (int i = 0; i < 8; ++i) cvt4(dst + 64 + i * 4, ys[i]);
#pragma unroll
      for (int j = 0; j < 32; ++j) dst[96 + j] = (_Float16)0.f;
    }
  }
  const int m16 = lane & 15, hi = lane >> 4, mb = hi * 8;
  gemm1_relu(&As[m16][0], &Hs[0][0], Wg1p, bg1, lane);
  v8f acc[2];
  gemm2_both(&Hs[m16][0], Wg2p, bg2, lane, &acc[0], &acc[1]);
#pragma unroll
  for (int nt = 0; nt < 2; ++nt) {
    const int n = nt * 16 + m16;
#pragma unroll
    for (int r = 0; r < 8; ++r) y[(size_t)(mb + r) * FD + n] = acc[nt][r];
  }
}

// ---------------------------------------------------------------------------
extern "C" void kernel_launch(void* const* d_in, const int* in_sizes, int n_in,
                              void* d_out, int out_size, void* d_ws, size_t ws_size,
                              hipStream_t stream) {
  (void)in_sizes; (void)n_in; (void)out_size; (void)ws_size;
  const float* x_in  = (const float*)d_in[0];
  const float* e_in  = (const float*)d_in[1];
  const float* y_in  = (const float*)d_in[2];
  const float* We1 = (const float*)d_in[3];  const float* be1 = (const float*)d_in[4];
  const float* We2 = (const float*)d_in[5];  const float* be2 = (const float*)d_in[6];
  const float* Wn1 = (const float*)d_in[7];  const float* bn1 = (const float*)d_in[8];
  const float* Wn2 = (const float*)d_in[9];  const float* bn2 = (const float*)d_in[10];
  const float* Wg1 = (const float*)d_in[11]; const float* bg1 = (const float*)d_in[12];
  const float* Wg2 = (const float*)d_in[13]; const float* bg2 = (const float*)d_in[14];
  const int* eidx = (const int*)d_in[15];
  const int* rowI = eidx;
  const int* colI = eidx + E_TOT;

  // State lives in d_out: [x (N,F) | edge_attr (E,F) | y (B,F)], updated in place.
  float* x_st = (float*)d_out;
  float* e_st = x_st + (size_t)N_NODES * FD;
  float* y_st = e_st + (size_t)E_TOT * FD;

  // Workspace layout
  float* e_sum_node  = (float*)d_ws;                       // N*F
  float* e_sum_graph = e_sum_node + (size_t)N_NODES * FD;  // B*F
  float* x_sum_graph = e_sum_graph + BG * FD;              // B*F
  float* deg         = x_sum_graph + BG * FD;              // N
  size_t hoff = (((size_t)N_NODES * FD + 2 * BG * FD + N_NODES) * 4 + 63) & ~(size_t)63;
  _Float16* hb   = (_Float16*)((char*)d_ws + hoff);
  _Float16* We1p = hb;            // 4*4*512 = 8192 halves
  _Float16* Wn1p = hb + 8192;
  _Float16* Wg1p = hb + 16384;
  _Float16* We2p = hb + 24576;    // 2*2*512 = 2048 halves
  _Float16* Wn2p = hb + 26624;
  _Float16* Wg2p = hb + 28672;

  hipMemcpyAsync(x_st, x_in, (size_t)N_NODES * FD * 4, hipMemcpyDeviceToDevice, stream);
  hipMemcpyAsync(e_st, e_in, (size_t)E_TOT * FD * 4, hipMemcpyDeviceToDevice, stream);
  hipMemcpyAsync(y_st, y_in, (size_t)BG * FD * 4, hipMemcpyDeviceToDevice, stream);

  zero_kernel<<<(N_NODES + 255) / 256, 256, 0, stream>>>(deg, N_NODES);
  degree_kernel<<<(E_TOT + 255) / 256, 256, 0, stream>>>(colI, deg);

  pack_b_kernel<<<32, 256, 0, stream>>>(We1, We1p, 128, HD, 4);
  pack_b_kernel<<<32, 256, 0, stream>>>(Wn1, Wn1p, 96, HD, 4);
  pack_b_kernel<<<32, 256, 0, stream>>>(Wg1, Wg1p, 96, HD, 4);
  pack_b_kernel<<<8, 256, 0, stream>>>(We2, We2p, HD, FD, 2);
  pack_b_kernel<<<8, 256, 0, stream>>>(Wn2, Wn2p, HD, FD, 2);
  pack_b_kernel<<<8, 256, 0, stream>>>(Wg2, Wg2p, HD, FD, 2);

  const int sums_len = N_NODES * FD + 2 * BG * FD;
  for (int it = 0; it < 5; ++it) {
    zero_kernel<<<(sums_len + 255) / 256, 256, 0, stream>>>(e_sum_node, sums_len);
    edge_kernel<<<E_TOT / 128, 256, 0, stream>>>(x_st, e_st, y_st, rowI, colI,
                                                 We1p, be1, We2p, be2,
                                                 e_sum_node, e_sum_graph);
    node_kernel<<<(N_NODES + 127) / 128, 256, 0, stream>>>(x_st, y_st, e_sum_node, deg,
                                                           Wn1p, bn1, Wn2p, bn2,
                                                           x_sum_graph);
    global_kernel<<<1, 32, 0, stream>>>(y_st, x_sum_graph, e_sum_graph,
                                        Wg1p, bg1, Wg2p, bg2);
  }
}
